// EnvEncoder_72249939853815
// MI455X (gfx1250) — matrix operand order repaired and checked by reading, not compiled, not run on hardware
//
#include <hip/hip_runtime.h>
#include <hip/hip_bf16.h>

typedef __attribute__((ext_vector_type(16))) _Float16 v16h;
typedef __attribute__((ext_vector_type(8)))  _Float16 v8h;
typedef __attribute__((ext_vector_type(2)))  _Float16 h2;
typedef __attribute__((ext_vector_type(8)))  float    v8f;

#define LN_EPS 1e-5f

// Workspace layout:
//   [0, 40960)      B : wsw  — fuse B-fragments, 8*5*32*16 f16
//   [40960, 71680)  B : wbd  — branch block-diag B-fragments, 10*3*32*16 f16
//   [71680, ...)      : cvec — 864 f32: bbr[160]|ln1_g[160]|ln1_b[160]|b_fuse[128]|ln2_g[128]|ln2_b[128]
#define WSW_N   (8 * 5 * 32 * 16)    // 20480
#define WBD_N   (10 * 3 * 32 * 16)   // 15360
#define FRAG_N  (WSW_N + WBD_N)      // 35840 f16
#define CVEC_N  864
#define PREP_N  (FRAG_N + CVEC_N)    // 36704 work items

// Cross-lane xor-butterfly within 16-lane halves: single ds_swizzle_b32.
// Group-of-32 mode: new_lane = ((lane & and_mask) | or_mask) ^ xor_mask
#define XSWZ(v, m) \
  __int_as_float(__builtin_amdgcn_ds_swizzle(__float_as_int(v), (((m) << 10) | 0x1f)))

// f32 pair -> packed f16 (RTZ), reinterpreted as our h2 type
__device__ __forceinline__ h2 pk_f16(float a, float b) {
  auto p = __builtin_amdgcn_cvt_pkrtz(a, b);
  return __builtin_bit_cast(h2, p);
}

// ---- One-shot prep: build swizzled f16 fragment images in d_ws -----------
__global__ __launch_bounds__(256) void envenc_prep(
    const float* __restrict__ w_month, const float* __restrict__ b_month,
    const float* __restrict__ w_area,  const float* __restrict__ b_area,
    const float* __restrict__ w_icls,  const float* __restrict__ b_icls,
    const float* __restrict__ w_scalar,const float* __restrict__ b_scalar,
    const float* __restrict__ w_long,  const float* __restrict__ b_long,
    const float* __restrict__ w_lat,   const float* __restrict__ b_lat,
    const float* __restrict__ w_hist,  const float* __restrict__ b_hist,
    const float* __restrict__ ln1_g,   const float* __restrict__ ln1_b,
    const float* __restrict__ w_fuse,  const float* __restrict__ b_fuse,
    const float* __restrict__ ln2_g,   const float* __restrict__ ln2_b,
    _Float16* __restrict__ ws)
{
  int i = blockIdx.x * 256 + threadIdx.x;
  if (i >= PREP_N) return;

  if (i < WSW_N) {
    // wsw[(((t*5)+c)*32+lane)*16+e] = w_fuse[(32c+8*(lane/16)+(e&7)+16*(e>>3))*128 + 16t+lane%16]
    int t    = i / 2560;
    int rem  = i - t * 2560;
    int c    = rem >> 9;
    int rem2 = rem & 511;
    int ln   = rem2 >> 4;
    int e    = rem2 & 15;
    int k    = 32 * c + 8 * (ln >> 4) + (e & 7) + 16 * (e >> 3);
    int n    = 16 * t + (ln & 15);
    ws[i] = (_Float16)w_fuse[k * 128 + n];
  } else if (i < FRAG_N) {
    int j    = i - WSW_N;
    int t    = j / 1536;
    int rem  = j - t * 1536;
    int c    = rem >> 9;
    int rem2 = rem & 511;
    int ln   = rem2 >> 4;
    int e    = rem2 & 15;
    int k    = 32 * c + 8 * (ln >> 4) + (e & 7) + 16 * (e >> 3);   // 0..95 (94,95 zero pad)
    int n    = 16 * t + (ln & 15);            // 0..159
    float v = 0.0f;
    if (k < 94) {
      if (n < 32)       { if (k < 12)            v = w_month[k * 32 + n]; }
      else if (n < 48)  { if (k >= 12 && k < 18) v = w_area[(k - 12) * 16 + (n - 32)]; }
      else if (n < 64)  { if (k >= 18 && k < 24) v = w_icls[(k - 18) * 16 + (n - 48)]; }
      else if (n < 80)  { if (k >= 24 && k < 26) v = w_scalar[(k - 24) * 16 + (n - 64)]; }
      else if (n < 112) { if (k >= 26 && k < 62) v = w_long[(k - 26) * 32 + (n - 80)]; }
      else if (n < 128) { if (k >= 62 && k < 74) v = w_lat[(k - 62) * 16 + (n - 112)]; }
      else              { if (k >= 74)           v = w_hist[(k - 74) * 32 + (n - 128)]; }
    }
    ws[i] = (_Float16)v;
  } else {
    int j = i - FRAG_N;                       // 0..863
    float* cvec = (float*)(ws + FRAG_N);
    float v;
    if (j < 160) {                            // concatenated branch biases
      if (j < 32)       v = b_month[j];
      else if (j < 48)  v = b_area[j - 32];
      else if (j < 64)  v = b_icls[j - 48];
      else if (j < 80)  v = b_scalar[j - 64];
      else if (j < 112) v = b_long[j - 80];
      else if (j < 128) v = b_lat[j - 112];
      else              v = b_hist[j - 128];
    } else if (j < 320) v = ln1_g[j - 160];
    else if (j < 480)   v = ln1_b[j - 320];
    else if (j < 608)   v = b_fuse[j - 480];
    else if (j < 736)   v = ln2_g[j - 608];
    else                v = ln2_b[j - 736];
    cvec[j] = v;
  }
}

// ---- Main kernel ---------------------------------------------------------
__global__ __launch_bounds__(256) void envenc_kernel(
    const float* __restrict__ x,
    const _Float16* __restrict__ ws,          // pre-swizzled fragments + cvec
    float* __restrict__ out)
{
  __shared__ __align__(32) _Float16 wfrag[FRAG_N];  // wsw | wbd  (71680 B)
  __shared__ float cvec[CVEC_N];                    // consts (3456 B)
  // Per-wave union block (32 rows * 160 f16 = 10240 B per wave):
  //   first x staging (stride 96 f16), then h staging (stride 160 f16)
  __shared__ __align__(16) _Float16 uni[256 * 160]; // 81920 B

  const int tid     = threadIdx.x;
  const int wv      = tid >> 5;
  const int lane    = tid & 31;
  const int halfsel = lane >> 4;
  const int lr      = lane & 15;
  _Float16* wblk    = &uni[wv * 5120];

  const _Float16* wsw  = wfrag;
  const _Float16* wbd  = wfrag + WSW_N;
  const float*    bbr  = cvec;
  const float*    g1   = cvec + 160;
  const float*    b1   = cvec + 320;
  const float*    bfu  = cvec + 480;
  const float*    g2   = cvec + 608;
  const float*    b2   = cvec + 736;

  // ---- Stage 0: straight contiguous copy of pre-swizzled images ----------
  {
    const uint4* src = (const uint4*)ws;
    uint4*       dst = (uint4*)wfrag;
#pragma unroll 2
    for (int i = tid; i < FRAG_N / 8; i += 256)     // 4480 16B chunks
      dst[i] = src[i];
    const float* csrc = (const float*)(ws + FRAG_N);
    for (int i = tid; i < CVEC_N; i += 256)
      cvec[i] = csrc[i];
  }

  // ---- Stage 1: x row -> LDS f16 (stride 96, cols 94/95 zero) ------------
  {
    const long long row = (long long)blockIdx.x * 256 + tid;
    const float2* xp2 = (const float2*)(x + row * 94);   // 376 B stride => 8B aligned
    _Float16* xrw = wblk + lane * 96;
#pragma unroll
    for (int jj = 0; jj < 11; ++jj) {                    // elems 0..87, 16B stores
      v8h chunk;
#pragma unroll
      for (int u = 0; u < 4; ++u) {
        float2 ab = xp2[4 * jj + u];
        h2 p = pk_f16(ab.x, ab.y);
        chunk[2 * u]     = p[0];
        chunk[2 * u + 1] = p[1];
      }
      *(v8h*)(xrw + 8 * jj) = chunk;
    }
#pragma unroll
    for (int u = 0; u < 3; ++u) {                        // elems 88..93
      float2 ab = xp2[44 + u];
      *(h2*)(xrw + 88 + 2 * u) = pk_f16(ab.x, ab.y);
    }
    h2 z; z[0] = (_Float16)0.0f; z[1] = (_Float16)0.0f;
    *(h2*)(xrw + 94) = z;                                // zero pad 94,95
  }
  __syncthreads();

  // ---- Stage 2: branch layer as block-diag GEMM (WMMA) + LN1 -------------
  // Hoist ALL x A-fragments (both 16-row tiles) before h overwrites x.
  v16h axf[2][3];
#pragma unroll
  for (int mt = 0; mt < 2; ++mt) {
#pragma unroll
    for (int c = 0; c < 3; ++c) {
      const _Float16* p = wblk + (mt * 16 + lr) * 96 + 32 * c + 8 * halfsel;
      v8h lo = *(const v8h*)p;
      v8h hi = *(const v8h*)(p + 16);
      v16h a;
#pragma unroll
      for (int e = 0; e < 8; ++e) { a[e] = lo[e]; a[e + 8] = hi[e]; }
      axf[mt][c] = a;
    }
  }

#pragma unroll
  for (int mt = 0; mt < 2; ++mt) {
    v8f acc[10];
#pragma unroll
    for (int t = 0; t < 10; ++t)
#pragma unroll
      for (int r = 0; r < 8; ++r) acc[t][r] = 0.0f;

    v16h bcur[3], bnxt[3];
#pragma unroll
    for (int c = 0; c < 3; ++c)
      bcur[c] = *(const v16h*)(&wbd[((0 * 3 + c) * 32 + lane) * 16]);
#pragma unroll
    for (int t = 0; t < 10; ++t) {
      if (t < 9) {
#pragma unroll
        for (int c = 0; c < 3; ++c)
          bnxt[c] = *(const v16h*)(&wbd[(((t + 1) * 3 + c) * 32 + lane) * 16]);
      }
#pragma unroll
      for (int c = 0; c < 3; ++c)
        acc[t] = __builtin_amdgcn_wmma_f32_16x16x32_f16(
            false, axf[mt][c], false, bcur[c], (short)0, acc[t], false, false);
#pragma unroll
      for (int c = 0; c < 3; ++c) bcur[c] = bnxt[c];
    }

    // bias + ReLU, LN1 stats over 160 via half-wave xor-swizzles
    float p_[8], q_[8];
#pragma unroll
    for (int r = 0; r < 8; ++r) { p_[r] = 0.0f; q_[r] = 0.0f; }
#pragma unroll
    for (int t = 0; t < 10; ++t) {
      float bn = bbr[16 * t + lr];
#pragma unroll
      for (int r = 0; r < 8; ++r) {
        float v = fmaxf(acc[t][r] + bn, 0.0f);
        acc[t][r] = v;
        p_[r] += v;
        q_[r] += v * v;
      }
    }
#pragma unroll
    for (int r = 0; r < 8; ++r) {
      p_[r] += XSWZ(p_[r], 1);
      p_[r] += XSWZ(p_[r], 2);
      p_[r] += XSWZ(p_[r], 4);
      p_[r] += XSWZ(p_[r], 8);
      q_[r] += XSWZ(q_[r], 1);
      q_[r] += XSWZ(q_[r], 2);
      q_[r] += XSWZ(q_[r], 4);
      q_[r] += XSWZ(q_[r], 8);
      float mu  = p_[r] * (1.0f / 160.0f);
      float var = q_[r] * (1.0f / 160.0f) - mu * mu;
      p_[r] = mu;
      q_[r] = rsqrtf(var + LN_EPS);
    }
    // LN1 + ReLU, scatter h (f16) into A-fragment rows (stride 160)
#pragma unroll
    for (int t = 0; t < 10; ++t) {
      int   n  = 16 * t + lr;
      float gg = g1[n];
      float bb = b1[n];
#pragma unroll
      for (int r = 0; r < 8; ++r) {
        float hv = fmaxf((acc[t][r] - p_[r]) * q_[r] * gg + bb, 0.0f);
        wblk[(mt * 16 + r + 8 * halfsel) * 160 + n] = (_Float16)hv;
      }
    }
  }
  __syncthreads();

  // ---- Stage 3: fuse GEMM (WMMA) + LN2 + ReLU + store --------------------
#pragma unroll
  for (int mt = 0; mt < 2; ++mt) {
    const _Float16* hA = wblk + (mt * 16 + lr) * 160 + 8 * halfsel;
    v16h afr[5];
#pragma unroll
    for (int c = 0; c < 5; ++c) {
      v8h lo = *(const v8h*)(hA + 32 * c);
      v8h hi = *(const v8h*)(hA + 32 * c + 16);
      v16h a;
#pragma unroll
      for (int e = 0; e < 8; ++e) { a[e] = lo[e]; a[e + 8] = hi[e]; }
      afr[c] = a;
    }

    v8f acc[8];
#pragma unroll
    for (int t = 0; t < 8; ++t)
#pragma unroll
      for (int r = 0; r < 8; ++r) acc[t][r] = 0.0f;

    v16h bcur[5], bnxt[5];
#pragma unroll
    for (int c = 0; c < 5; ++c)
      bcur[c] = *(const v16h*)(&wsw[((0 * 5 + c) * 32 + lane) * 16]);
#pragma unroll
    for (int t = 0; t < 8; ++t) {
      if (t < 7) {
#pragma unroll
        for (int c = 0; c < 5; ++c)
          bnxt[c] = *(const v16h*)(&wsw[(((t + 1) * 5 + c) * 32 + lane) * 16]);
      }
#pragma unroll
      for (int c = 0; c < 5; ++c)
        acc[t] = __builtin_amdgcn_wmma_f32_16x16x32_f16(
            false, afr[c], false, bcur[c], (short)0, acc[t], false, false);
#pragma unroll
      for (int c = 0; c < 5; ++c) bcur[c] = bnxt[c];
    }

    // + b_fuse, LN2 stats via half-wave xor-swizzles, ReLU, store
    float p_[8], q_[8];
#pragma unroll
    for (int r = 0; r < 8; ++r) { p_[r] = 0.0f; q_[r] = 0.0f; }
#pragma unroll
    for (int t = 0; t < 8; ++t) {
      float bn = bfu[16 * t + lr];
#pragma unroll
      for (int r = 0; r < 8; ++r) {
        float d = acc[t][r] + bn;
        acc[t][r] = d;
        p_[r] += d;
        q_[r] += d * d;
      }
    }
#pragma unroll
    for (int r = 0; r < 8; ++r) {
      p_[r] += XSWZ(p_[r], 1);
      p_[r] += XSWZ(p_[r], 2);
      p_[r] += XSWZ(p_[r], 4);
      p_[r] += XSWZ(p_[r], 8);
      q_[r] += XSWZ(q_[r], 1);
      q_[r] += XSWZ(q_[r], 2);
      q_[r] += XSWZ(q_[r], 4);
      q_[r] += XSWZ(q_[r], 8);
      float mean = p_[r] * (1.0f / 128.0f);
      float var  = q_[r] * (1.0f / 128.0f) - mean * mean;
      p_[r] = mean;
      q_[r] = rsqrtf(var + LN_EPS);
    }

    const long long rbase = (long long)blockIdx.x * 256 + wv * 32 + mt * 16 + 8 * halfsel;
#pragma unroll
    for (int t = 0; t < 8; ++t) {
      int   n   = 16 * t + lr;
      float g2n = g2[n];
      float b2n = b2[n];
#pragma unroll
      for (int r = 0; r < 8; ++r) {
        float v = fmaxf((acc[t][r] - p_[r]) * q_[r] * g2n + b2n, 0.0f);
        out[(rbase + r) * 128 + n] = v;
      }
    }
  }
}

extern "C" void kernel_launch(void* const* d_in, const int* in_sizes, int n_in,
                              void* d_out, int out_size, void* d_ws, size_t ws_size,
                              hipStream_t stream) {
  (void)n_in; (void)out_size; (void)ws_size;
  const float* x        = (const float*)d_in[0];
  const float* w_month  = (const float*)d_in[1];
  const float* b_month  = (const float*)d_in[2];
  const float* w_area   = (const float*)d_in[3];
  const float* b_area   = (const float*)d_in[4];
  const float* w_icls   = (const float*)d_in[5];
  const float* b_icls   = (const float*)d_in[6];
  const float* w_scalar = (const float*)d_in[7];
  const float* b_scalar = (const float*)d_in[8];
  const float* w_long   = (const float*)d_in[9];
  const float* b_long   = (const float*)d_in[10];
  const float* w_lat    = (const float*)d_in[11];
  const float* b_lat    = (const float*)d_in[12];
  const float* w_hist   = (const float*)d_in[13];
  const float* b_hist   = (const float*)d_in[14];
  const float* ln1_g    = (const float*)d_in[15];
  const float* ln1_b    = (const float*)d_in[16];
  const float* w_fuse   = (const float*)d_in[17];
  const float* b_fuse   = (const float*)d_in[18];
  const float* ln2_g    = (const float*)d_in[19];
  const float* ln2_b    = (const float*)d_in[20];
  float* out = (float*)d_out;
  _Float16* ws = (_Float16*)d_ws;

  // One-shot swizzle of weights/consts into workspace.
  envenc_prep<<<(PREP_N + 255) / 256, 256, 0, stream>>>(
      w_month, b_month, w_area, b_area, w_icls, b_icls, w_scalar, b_scalar,
      w_long, b_long, w_lat, b_lat, w_hist, b_hist, ln1_g, ln1_b,
      w_fuse, b_fuse, ln2_g, ln2_b, ws);

  const int nrows  = in_sizes[0] / 94;   // B = 524288
  const int blocks = nrows / 256;        // 2048 blocks of 256 rows

  envenc_kernel<<<blocks, 256, 0, stream>>>(x, ws, out);
}
